// Decoder_51376398794974
// MI455X (gfx1250) — matrix operand order repaired and checked by reading, not compiled, not run on hardware
//
#include <hip/hip_runtime.h>

#define BB 64
#define SS 2048
#define EE 1024
#define HH 512
#define EMB 256
#define VV 50257
#define XK 1792  // E + EMB + H = 1024+256+512

typedef __attribute__((ext_vector_type(16))) __bf16 v16bf;
typedef __attribute__((ext_vector_type(8)))  float  v8f;

union Frag16 { v16bf bf; unsigned int d[8]; unsigned short u[16]; };

__device__ __forceinline__ unsigned short f2bf(float f) {
  unsigned int x = __builtin_bit_cast(unsigned int, f);
  x += 0x7fffu + ((x >> 16) & 1u);
  return (unsigned short)(x >> 16);
}

__device__ __forceinline__ float warp_sum(float v) {
  for (int m = 1; m < 32; m <<= 1) v += __shfl_xor(v, m, 32);
  return v;
}
__device__ __forceinline__ float warp_max(float v) {
  for (int m = 1; m < 32; m <<= 1) v = fmaxf(v, __shfl_xor(v, m, 32));
  return v;
}
__device__ __forceinline__ float sigm(float x) { return 1.f / (1.f + expf(-x)); }

// gfx1250 async DMA: global -> LDS, 16 bytes per lane, tracked by ASYNCcnt.
// LDS address = low 32 bits of the flat shared pointer (ISA 10.2: LDS aperture
// truncates to wave-relative LDS offset).
__device__ __forceinline__ void async_g2l_b128(unsigned lds_addr, const void* gaddr) {
  asm volatile("global_load_async_to_lds_b128 %0, %1, off"
               :: "v"(lds_addr), "v"(gaddr) : "memory");
}
__device__ __forceinline__ void wait_asynccnt0() {
  asm volatile("s_wait_asynccnt 0" ::: "memory");
}
__device__ __forceinline__ unsigned lds_lo32(const void* p) {
  return (unsigned)(size_t)p;
}

// ---------------------------------------------------------------- f32 -> bf16
__global__ __launch_bounds__(256) void k_cvt_bf16(const float* __restrict__ src,
                                                  unsigned short* __restrict__ dst, int n) {
  int i = blockIdx.x * 256 + threadIdx.x;
  if (i < n) dst[i] = f2bf(src[i]);
}

// ---------------------------------------------------------------- ws_app = h0 @ Wsᵀ + b
__global__ __launch_bounds__(256) void k_wsapp(const float* __restrict__ h0,
                                               const float* __restrict__ wsw,
                                               const float* __restrict__ wsb,
                                               float* __restrict__ wsapp) {
  int idx = blockIdx.x * 256 + threadIdx.x;      // B*H = 32768
  int b = idx >> 9, n = idx & 511;
  const float* hr = h0 + b * HH;
  const float* wr = wsw + (size_t)n * HH;
  float s = wsb[n];
  for (int k = 0; k < HH; ++k) s += hr[k] * wr[k];
  wsapp[idx] = s;
}

// ---------------------------------------------------------------- fused energy GEMM
// energy[m] = sum_n tanh( enc_out[m,:] @ W[n,:] + wh_b[n] + ws_app[b,n] ) * attn_v[n]
__global__ __launch_bounds__(256) void k_energy(const float* __restrict__ enc,
                                                const unsigned short* __restrict__ wbf,
                                                const float* __restrict__ whb,
                                                const float* __restrict__ wsapp,
                                                const float* __restrict__ av,
                                                float* __restrict__ energy) {
  __shared__ unsigned short Ald[32][40];
  __shared__ unsigned short Bld[512][40];
  __shared__ float ered[32];

  int tid = threadIdx.x;
  int lane = tid & 31, w = tid >> 5;
  int half = lane >> 4, l15 = lane & 15;
  int mrow = w >> 2, nq = w & 3;             // wave -> (m-tile, N quarter of 8 tiles)
  int m0 = blockIdx.x * 32;
  int bblk = m0 >> 11;                       // 32 | 2048 => one batch per block

  unsigned bld0 = lds_lo32(&Bld[0][0]);

  v8f acc[8];
  for (int it = 0; it < 8; ++it)
    for (int r = 0; r < 8; ++r) acc[it][r] = 0.f;

  for (int k0 = 0; k0 < EE; k0 += 32) {
    // B tile 512x32 bf16: async DMA global->LDS (8 x b128 per lane)
    for (int i = 0; i < 8; ++i) {
      int idx = tid + i * 256;
      int n = idx >> 2, kp4 = idx & 3;
      async_g2l_b128(bld0 + (unsigned)(n * 80 + kp4 * 16),
                     wbf + (size_t)n * EE + k0 + kp4 * 8);
    }
    // A tile 32x32 f32 -> bf16 (one float4 per thread, needs conversion)
    {
      int row = tid >> 3, kq = tid & 7;
      float4 f = *(const float4*)(enc + (size_t)(m0 + row) * EE + k0 + kq * 4);
      Ald[row][kq * 4 + 0] = f2bf(f.x);
      Ald[row][kq * 4 + 1] = f2bf(f.y);
      Ald[row][kq * 4 + 2] = f2bf(f.z);
      Ald[row][kq * 4 + 3] = f2bf(f.w);
    }
    wait_asynccnt0();
    __syncthreads();

    Frag16 a;
    for (int p = 0; p < 8; ++p) {
      int kb = ((p < 4) ? 2 * p : 8 + 2 * p) + 8 * half;
      a.d[p] = *(const unsigned int*)&Ald[mrow * 16 + l15][kb];
    }
    for (int it = 0; it < 8; ++it) {
      Frag16 bfr;
      int n = (nq * 8 + it) * 16 + l15;
      for (int p = 0; p < 8; ++p)
        bfr.d[p] = *(const unsigned int*)&Bld[n][((p < 4) ? 2 * p : 8 + 2 * p) + 8 * half];
      acc[it] = __builtin_amdgcn_wmma_f32_16x16x32_bf16(
          false, a.bf, false, bfr.bf, (short)0, acc[it], false, false);
    }
    __syncthreads();
  }

  // fused epilogue: tanh + dot(attn_v) + row reduction
  if (tid < 32) ered[tid] = 0.f;
  __syncthreads();

  float rsum[8];
  for (int r = 0; r < 8; ++r) rsum[r] = 0.f;
  for (int it = 0; it < 8; ++it) {
    int n = (nq * 8 + it) * 16 + l15;
    float bias = whb[n] + wsapp[bblk * HH + n];
    float vav = av[n];
    for (int r = 0; r < 8; ++r)
      rsum[r] += tanhf(acc[it][r] + bias) * vav;
  }
  for (int r = 0; r < 8; ++r) {
    float s = rsum[r];
    s += __shfl_xor(s, 1, 32);
    s += __shfl_xor(s, 2, 32);
    s += __shfl_xor(s, 4, 32);
    s += __shfl_xor(s, 8, 32);
    if (l15 == 0) atomicAdd(&ered[mrow * 16 + r + 8 * half], s);
  }
  __syncthreads();
  if (tid < 32) energy[m0 + tid] = ered[tid];
}

// ---------------------------------------------------------------- softmax over S (in place)
__global__ __launch_bounds__(256) void k_softmax_s(float* __restrict__ e) {
  int b = blockIdx.x, tid = threadIdx.x;
  __shared__ float red[8];
  __shared__ float bc;
  float* row = e + (size_t)b * SS;
  float mx = -3.4e38f;
  for (int s = tid; s < SS; s += 256) mx = fmaxf(mx, row[s]);
  mx = warp_max(mx);
  if ((tid & 31) == 0) red[tid >> 5] = mx;
  __syncthreads();
  if (tid == 0) { float m = red[0]; for (int i = 1; i < 8; ++i) m = fmaxf(m, red[i]); bc = m; }
  __syncthreads();
  mx = bc;
  float sum = 0.f;
  for (int s = tid; s < SS; s += 256) { float v = expf(row[s] - mx); row[s] = v; sum += v; }
  sum = warp_sum(sum);
  __syncthreads();
  if ((tid & 31) == 0) red[tid >> 5] = sum;
  __syncthreads();
  if (tid == 0) { float t = 0; for (int i = 0; i < 8; ++i) t += red[i]; bc = t; }
  __syncthreads();
  float inv = 1.f / bc;
  for (int s = tid; s < SS; s += 256) row[s] *= inv;
}

// ---------------------------------------------------------------- context[b,e] = attn[b,:] @ enc[b,:,e]
__global__ __launch_bounds__(256) void k_context(const float* __restrict__ attn,
                                                 const float* __restrict__ enc,
                                                 float* __restrict__ ctx) {
  int b = blockIdx.y;
  int e = blockIdx.x * 256 + threadIdx.x;    // grid.x = 4
  const float* ar = attn + (size_t)b * SS;
  const float* er = enc + (size_t)b * SS * EE + e;
  float s = 0.f;
  for (int t = 0; t < SS; ++t) s += ar[t] * er[(size_t)t * EE];
  ctx[b * EE + e] = s;
}

// ---------------------------------------------------------------- xcat = bf16([context|embed|h0]); also embed f32
__global__ __launch_bounds__(256) void k_xcat(const float* __restrict__ ctx,
                                              const float* __restrict__ embtab,
                                              const int* __restrict__ decin,
                                              const float* __restrict__ h0,
                                              unsigned short* __restrict__ xcat,
                                              float* __restrict__ embf) {
  int idx = blockIdx.x * 256 + threadIdx.x;  // B*1792 = 114688 (exact)
  int b = idx / XK, k = idx % XK;
  float v;
  if (k < EE) v = ctx[b * EE + k];
  else if (k < EE + EMB) {
    v = embtab[(size_t)decin[b] * EMB + (k - EE)];
    embf[b * EMB + (k - EE)] = v;
  } else v = h0[b * HH + (k - EE - EMB)];
  xcat[idx] = f2bf(v);
}

// ---------------------------------------------------------------- gates = xcat @ [Wih|Whh]ᵀ + biases (M=64,N=2048,K=1792)
__global__ __launch_bounds__(256) void k_gates(const unsigned short* __restrict__ xcat,
                                               const float* __restrict__ wih,
                                               const float* __restrict__ whh,
                                               const float* __restrict__ bih,
                                               const float* __restrict__ bhh,
                                               float* __restrict__ gates) {
  __shared__ unsigned short Axc[64][40];
  __shared__ unsigned short Bw[32][40];
  int tid = threadIdx.x, lane = tid & 31, w = tid >> 5;
  int half = lane >> 4, l15 = lane & 15;
  int mt = w & 3, nt = w >> 2;               // 4 M-tiles x 2 N-tiles
  int n0 = blockIdx.x * 32;
  unsigned axc0 = lds_lo32(&Axc[0][0]);

  v8f acc;
  for (int r = 0; r < 8; ++r) acc[r] = 0.f;

  for (int k0 = 0; k0 < XK; k0 += 32) {
    {                                        // A: 64x32 bf16 async DMA (one b128 per lane)
      int row = tid >> 2, kp4 = tid & 3;
      async_g2l_b128(axc0 + (unsigned)(row * 80 + kp4 * 16),
                     xcat + (size_t)row * XK + k0 + kp4 * 8);
    }
    {                                        // B: 32x32 f32 -> bf16 (fused Wih|Whh)
      int nloc = tid >> 3, kq = tid & 7;
      int n = n0 + nloc, k = k0 + kq * 4;
      float4 f;
      if (k < EE + EMB) f = *(const float4*)(wih + (size_t)n * (EE + EMB) + k);
      else              f = *(const float4*)(whh + (size_t)n * HH + (k - EE - EMB));
      Bw[nloc][kq * 4 + 0] = f2bf(f.x);
      Bw[nloc][kq * 4 + 1] = f2bf(f.y);
      Bw[nloc][kq * 4 + 2] = f2bf(f.z);
      Bw[nloc][kq * 4 + 3] = f2bf(f.w);
    }
    wait_asynccnt0();
    __syncthreads();
    Frag16 a, bfr;
    for (int p = 0; p < 8; ++p) {
      int kb = ((p < 4) ? 2 * p : 8 + 2 * p) + 8 * half;
      a.d[p]   = *(const unsigned int*)&Axc[mt * 16 + l15][kb];
      bfr.d[p] = *(const unsigned int*)&Bw[nt * 16 + l15][kb];
    }
    acc = __builtin_amdgcn_wmma_f32_16x16x32_bf16(
        false, a.bf, false, bfr.bf, (short)0, acc, false, false);
    __syncthreads();
  }
  int n = n0 + nt * 16 + l15;
  float bias = bih[n] + bhh[n];
  for (int r = 0; r < 8; ++r) {
    int bb = mt * 16 + r + 8 * half;
    gates[(size_t)bb * (4 * HH) + n] = acc[r] + bias;
  }
}

// ---------------------------------------------------------------- LSTM cell
__global__ __launch_bounds__(256) void k_lstm(const float* __restrict__ gates,
                                              const float* __restrict__ c0,
                                              float* __restrict__ out,
                                              float* __restrict__ htf,
                                              unsigned short* __restrict__ ht16) {
  int idx = blockIdx.x * 256 + threadIdx.x;  // B*H = 32768
  int b = idx >> 9, h = idx & 511;
  const float* g = gates + (size_t)b * (4 * HH);
  float gi = g[h], gf = g[HH + h], gg = g[2 * HH + h], go = g[3 * HH + h];
  float c = sigm(gf) * c0[idx] + sigm(gi) * tanhf(gg);
  float hh = sigm(go) * tanhf(c);
  out[(size_t)BB * VV + idx] = hh;                 // h_t
  out[(size_t)BB * VV + BB * HH + idx] = c;        // c_t
  htf[idx] = hh;
  ht16[idx] = f2bf(hh);
}

// ---------------------------------------------------------------- logits = h_t @ v_wᵀ + v_b (M=64,N=50257,K=512)
__global__ __launch_bounds__(256) void k_vocab(const unsigned short* __restrict__ ht16,
                                               const float* __restrict__ vw,
                                               const float* __restrict__ vb,
                                               float* __restrict__ logits) {
  __shared__ unsigned short Ah[64][40];
  __shared__ unsigned short Bw[32][40];
  int tid = threadIdx.x, lane = tid & 31, w = tid >> 5;
  int half = lane >> 4, l15 = lane & 15;
  int mt = w & 3, nt = w >> 2;
  int n0 = blockIdx.x * 32;
  unsigned ah0 = lds_lo32(&Ah[0][0]);

  v8f acc;
  for (int r = 0; r < 8; ++r) acc[r] = 0.f;

  for (int k0 = 0; k0 < HH; k0 += 32) {
    {                                        // A: 64x32 bf16 async DMA (one b128 per lane)
      int row = tid >> 2, kp4 = tid & 3;
      async_g2l_b128(ah0 + (unsigned)(row * 80 + kp4 * 16),
                     ht16 + (size_t)row * HH + k0 + kp4 * 8);
    }
    {                                        // B: 32x32 f32 -> bf16 with N guard
      int nloc = tid >> 3, kq = tid & 7;
      int n = n0 + nloc, k = k0 + kq * 4;
      float4 f = {0.f, 0.f, 0.f, 0.f};
      if (n < VV) f = *(const float4*)(vw + (size_t)n * HH + k);
      Bw[nloc][kq * 4 + 0] = f2bf(f.x);
      Bw[nloc][kq * 4 + 1] = f2bf(f.y);
      Bw[nloc][kq * 4 + 2] = f2bf(f.z);
      Bw[nloc][kq * 4 + 3] = f2bf(f.w);
    }
    wait_asynccnt0();
    __syncthreads();
    Frag16 a, bfr;
    for (int p = 0; p < 8; ++p) {
      int kb = ((p < 4) ? 2 * p : 8 + 2 * p) + 8 * half;
      a.d[p]   = *(const unsigned int*)&Ah[mt * 16 + l15][kb];
      bfr.d[p] = *(const unsigned int*)&Bw[nt * 16 + l15][kb];
    }
    acc = __builtin_amdgcn_wmma_f32_16x16x32_bf16(
        false, a.bf, false, bfr.bf, (short)0, acc, false, false);
    __syncthreads();
  }
  int n = n0 + nt * 16 + l15;
  if (n < VV) {
    float bias = vb[n];
    for (int r = 0; r < 8; ++r) {
      int bb = mt * 16 + r + 8 * half;
      logits[(size_t)bb * VV + n] = acc[r] + bias;
    }
  }
}

// ---------------------------------------------------------------- p_gen
__global__ __launch_bounds__(256) void k_pgen(const float* __restrict__ ctx,
                                              const float* __restrict__ htf,
                                              const float* __restrict__ embf,
                                              const float* __restrict__ whv,
                                              const float* __restrict__ wsv,
                                              const float* __restrict__ wxv,
                                              float* __restrict__ pgen) {
  int b = blockIdx.x, tid = threadIdx.x;
  __shared__ float red[8];
  float s = 0.f;
  for (int k = tid; k < EE; k += 256) s += ctx[b * EE + k] * whv[k];
  for (int k = tid; k < HH; k += 256) s += htf[b * HH + k] * wsv[k];
  for (int k = tid; k < EMB; k += 256) s += embf[b * EMB + k] * wxv[k];
  s = warp_sum(s);
  if ((tid & 31) == 0) red[tid >> 5] = s;
  __syncthreads();
  if (tid == 0) {
    float t = 0.f;
    for (int i = 0; i < 8; ++i) t += red[i];
    pgen[b] = sigm(t);
  }
}

// ---------------------------------------------------------------- out = p_gen * softmax_V(logits)
__global__ __launch_bounds__(256) void k_out_vocab(const float* __restrict__ logits,
                                                   const float* __restrict__ pgen,
                                                   float* __restrict__ out) {
  int b = blockIdx.x, tid = threadIdx.x;
  __shared__ float red[8];
  __shared__ float bc;
  const float* row = logits + (size_t)b * VV;
  float mx = -3.4e38f;
  for (int v = tid; v < VV; v += 256) mx = fmaxf(mx, row[v]);
  mx = warp_max(mx);
  if ((tid & 31) == 0) red[tid >> 5] = mx;
  __syncthreads();
  if (tid == 0) { float m = red[0]; for (int i = 1; i < 8; ++i) m = fmaxf(m, red[i]); bc = m; }
  __syncthreads();
  mx = bc;
  float sum = 0.f;
  for (int v = tid; v < VV; v += 256) sum += expf(row[v] - mx);
  sum = warp_sum(sum);
  __syncthreads();
  if ((tid & 31) == 0) red[tid >> 5] = sum;
  __syncthreads();
  if (tid == 0) { float t = 0.f; for (int i = 0; i < 8; ++i) t += red[i]; bc = t; }
  __syncthreads();
  float scale = pgen[b] / bc;
  float* orow = out + (size_t)b * VV;
  for (int v = tid; v < VV; v += 256) orow[v] = expf(row[v] - mx) * scale;
}

// ---------------------------------------------------------------- pointer scatter: out += (1-pg)*attn at enc_inputs
__global__ __launch_bounds__(256) void k_scatter(const float* __restrict__ attn,
                                                 const int* __restrict__ encin,
                                                 const float* __restrict__ pgen,
                                                 float* __restrict__ out) {
  int idx = blockIdx.x * 256 + threadIdx.x;  // B*S
  int b = idx >> 11;
  float val = (1.f - pgen[b]) * attn[idx];
  int v = encin[idx];
  atomicAdd(out + (size_t)b * VV + v, val);
}

// ---------------------------------------------------------------- launcher
extern "C" void kernel_launch(void* const* d_in, const int* in_sizes, int n_in,
                              void* d_out, int out_size, void* d_ws, size_t ws_size,
                              hipStream_t stream) {
  const float* enc_out     = (const float*)d_in[0];
  const float* h0          = (const float*)d_in[1];
  const float* c0          = (const float*)d_in[2];
  const int*   dec_input   = (const int*)d_in[3];
  const int*   enc_inputs  = (const int*)d_in[4];
  const float* embed_table = (const float*)d_in[5];
  const float* attn_wh_w   = (const float*)d_in[6];
  const float* attn_wh_b   = (const float*)d_in[7];
  const float* attn_ws_w   = (const float*)d_in[8];
  const float* attn_ws_b   = (const float*)d_in[9];
  const float* attn_v      = (const float*)d_in[10];
  const float* lstm_w_ih   = (const float*)d_in[11];
  const float* lstm_w_hh   = (const float*)d_in[12];
  const float* lstm_b_ih   = (const float*)d_in[13];
  const float* lstm_b_hh   = (const float*)d_in[14];
  const float* wh_vec      = (const float*)d_in[15];
  const float* ws_vec      = (const float*)d_in[16];
  const float* wx_vec      = (const float*)d_in[17];
  const float* v_w         = (const float*)d_in[18];
  const float* v_b         = (const float*)d_in[19];
  float* out = (float*)d_out;
  char*  ws  = (char*)d_ws;

  // workspace layout (bytes)
  size_t off = 0;
  float*          attn   = (float*)(ws + off);          off += (size_t)BB * SS * 4;   // energy -> attn
  float*          wsapp  = (float*)(ws + off);          off += (size_t)BB * HH * 4;
  float*          ctx    = (float*)(ws + off);          off += (size_t)BB * EE * 4;
  float*          embf   = (float*)(ws + off);          off += (size_t)BB * EMB * 4;
  unsigned short* xcat   = (unsigned short*)(ws + off); off += (size_t)BB * XK * 2;
  float*          gates  = (float*)(ws + off);          off += (size_t)BB * 4 * HH * 4;
  float*          htf    = (float*)(ws + off);          off += (size_t)BB * HH * 4;
  unsigned short* ht16   = (unsigned short*)(ws + off); off += (size_t)BB * HH * 2;
  float*          pgen   = (float*)(ws + off);          off += (size_t)BB * 4;
  float*          logits = (float*)(ws + off);          off += (size_t)BB * VV * 4;
  unsigned short* wbf    = (unsigned short*)(ws + off); off += (size_t)HH * EE * 2;

  k_cvt_bf16<<<(HH * EE) / 256, 256, 0, stream>>>(attn_wh_w, wbf, HH * EE);
  k_wsapp<<<(BB * HH) / 256, 256, 0, stream>>>(h0, attn_ws_w, attn_ws_b, wsapp);
  k_energy<<<(BB * SS) / 32, 256, 0, stream>>>(enc_out, wbf, attn_wh_b, wsapp, attn_v, attn);
  k_softmax_s<<<BB, 256, 0, stream>>>(attn);
  k_context<<<dim3(EE / 256, BB), 256, 0, stream>>>(attn, enc_out, ctx);
  k_xcat<<<(BB * XK) / 256, 256, 0, stream>>>(ctx, embed_table, dec_input, h0, xcat, embf);
  k_gates<<<(4 * HH) / 32, 256, 0, stream>>>(xcat, lstm_w_ih, lstm_w_hh, lstm_b_ih, lstm_b_hh, gates);
  k_lstm<<<(BB * HH) / 256, 256, 0, stream>>>(gates, c0, out, htf, ht16);
  k_vocab<<<(VV + 31) / 32, 256, 0, stream>>>(ht16, v_w, v_b, logits);
  k_pgen<<<BB, 256, 0, stream>>>(ctx, htf, embf, wh_vec, ws_vec, wx_vec, pgen);
  k_out_vocab<<<BB, 256, 0, stream>>>(logits, pgen, out);
  k_scatter<<<(BB * SS) / 256, 256, 0, stream>>>(attn, enc_inputs, pgen, out);
}
